// CharRNN_48129403519327
// MI455X (gfx1250) — matrix operand order repaired and checked by reading, not compiled, not run on hardware
//
#include <hip/hip_runtime.h>

typedef __attribute__((ext_vector_type(16))) _Float16 v16h;
typedef __attribute__((ext_vector_type(8)))  _Float16 v8h;
typedef __attribute__((ext_vector_type(8)))  float    v8f;
typedef __attribute__((ext_vector_type(4)))  float    v4f;

#define UNITS 65
#define BATCH 50
#define SEQ   8192
#define NGATE 260      // 4*UNITS
#define KP    64       // K via WMMA: k = 0..63 ; k = 64 handled as rank-1 FMA
#define NT_Z  17       // ceil(260/16) N tiles for the gate GEMM
#define NZC   288      // 18*16 z columns incl. 16 dummy cols (guard-free stores)
#define MT    16       // batch tile = WMMA M
#define NT_D  5        // ceil(65/16) N tiles for the dense projection
#define NTHREADS 288   // 9 waves: tiles nt = wid + 9*j, j in {0,1}

// Single-instruction transcendentals: v_exp_f32 + v_rcp_f32, no IEEE div chain.
__device__ __forceinline__ float fast_sigmoid(float x) {
    return __builtin_amdgcn_rcpf(1.0f + __expf(-x));
}
__device__ __forceinline__ float fast_tanh(float x) {
    // tanh(x) = 1 - 2/(exp(2x)+1); saturates correctly at +/-1.
    return 1.0f - 2.0f * __builtin_amdgcn_rcpf(1.0f + __expf(2.0f * x));
}

// B-matrix (K=32 x N=16) f16 fragment from a row-major global f32 matrix
// src[k][n] (leading dim ld, valid region kmax x nmax), zero padded.
// ISA layout: n = lane&15 ; K = (lane>>4)*8 + j (j<8), (lane>>4)*8 + 16 + (j-8) (j>=8).
__device__ __forceinline__ v16h load_bfrag_global(const float* src, int ld,
                                                  int kmax, int nmax,
                                                  int ntile, int kslice, int lane) {
    v16h r;
    const int n  = ntile * 16 + (lane & 15);
    const int kb = kslice * 32 + ((lane >> 4) << 3);
#pragma unroll
    for (int j = 0; j < 16; ++j) {
        const int k = kb + (j < 8 ? j : j + 8);
        float v = 0.0f;
        if (k < kmax && n < nmax) v = src[k * ld + n];
        r[j] = (_Float16)v;
    }
    return r;
}

// A-matrix (M=16 x K=32) f16 fragment from row-major LDS buffer, ld = KP.
// ISA layout: m = lane&15 ; elems 0..7: K = base+0..7 ; elems 8..15: K = base+16..23
// with base = kslice*32 + (lane>=16 ? 8 : 0). Two aligned 16B LDS reads.
__device__ __forceinline__ v16h load_afrag_lds(const _Float16* buf, int kslice, int lane) {
    const int m  = lane & 15;
    const int kb = kslice * 32 + ((lane >> 4) << 3);
    union { v16h v; v8h h[2]; } u;
    u.h[0] = *(const v8h*)(buf + m * KP + kb);
    u.h[1] = *(const v8h*)(buf + m * KP + kb + 16);
    return u.v;
}

// Load this lane's 8 C/D-rows (mb..mb+7) of a 16-float transposed column array.
__device__ __forceinline__ v8f load_col8(const float* col, int mb) {
    union { v8f v; v4f q[2]; } u;
    u.q[0] = *(const v4f*)(col + mb);
    u.q[1] = *(const v4f*)(col + mb + 4);
    return u.v;
}

// Accumulate one 16x16 gate tile: in@W + h@U + b + rank-1(k=64). No stores.
__device__ __forceinline__ v8f gate_tile_acc(const v16h (&Ain)[2], const v16h (&Ah)[2],
                                             const v16h (&bw)[2], const v16h (&bu)[2],
                                             float bv, float w64, float u64,
                                             const v8f& inT, const v8f& hT) {
    v8f accA = {bv, bv, bv, bv, bv, bv, bv, bv};           // bias folded into C
    v8f accB = {};
#pragma unroll
    for (int ks = 0; ks < 2; ++ks) {                       // two independent chains
        accA = __builtin_amdgcn_wmma_f32_16x16x32_f16(
            false, Ain[ks], false, bw[ks], (short)0, accA, false, false);
        accB = __builtin_amdgcn_wmma_f32_16x16x32_f16(
            false, Ah[ks], false, bu[ks], (short)0, accB, false, false);
    }
    v8f r = accA + accB;
#pragma unroll
    for (int v = 0; v < 8; ++v)                            // rank-1 k=64 term
        r[v] = __builtin_fmaf(inT[v], w64, __builtin_fmaf(hT[v], u64, r[v]));
    return r;
}

// One LSTM cell for a 16-row batch tile; z column-major [NZC][16].
// Both tile slots are computed and stored by every wave (wave 8's slot-1
// fragments are zero and its slot-1 columns land in the dummy z region),
// so there is no exec-mask manipulation anywhere in the WMMA phase.
__device__ __forceinline__ void lstm_cell(const _Float16* in_act, const float* in_col,
                                          _Float16* h_act, float* h_col,
                                          float* c_st, float* z,
                                          const v16h (&BW)[2][2], const v16h (&BU)[2][2],
                                          const float (&BB)[2],
                                          const float (&W64)[2], const float (&U64)[2],
                                          int wid, int lane, int m_e, int p_e) {
    v16h Ain[2], Ah[2];
#pragma unroll
    for (int ks = 0; ks < 2; ++ks) {
        Ain[ks] = load_afrag_lds(in_act, ks, lane);
        Ah[ks]  = load_afrag_lds(h_act,  ks, lane);
    }
    const int mb = (lane >> 4) << 3;
    const v8f inT = load_col8(in_col, mb);                 // in[row][64]
    const v8f hT  = load_col8(h_col,  mb);                 // h[row][64]

    union { v8f v; v4f q[2]; } z0, z1;                     // 4 independent WMMA chains
    z0.v = gate_tile_acc(Ain, Ah, BW[0], BU[0], BB[0], W64[0], U64[0], inT, hT);
    z1.v = gate_tile_acc(Ain, Ah, BW[1], BU[1], BB[1], W64[1], U64[1], inT, hT);

    const int n0 = wid * 16 + (lane & 15);                 // slot 0: nt = wid
    *(v4f*)(z + n0 * 16 + mb)               = z0.q[0];     // four ds_store_b128
    *(v4f*)(z + n0 * 16 + mb + 4)           = z0.q[1];
    *(v4f*)(z + (n0 + 144) * 16 + mb)       = z1.q[0];     // slot 1: nt = wid + 9
    *(v4f*)(z + (n0 + 144) * 16 + mb + 4)   = z1.q[1];
    __syncthreads();

    // Elementwise gates (first 256 threads), Keras order i,f,g,o.
    if (p_e < 16) {
#pragma unroll
        for (int r = 0; r < 4; ++r) {
            const int p  = p_e + 16 * r;
            const float gi = fast_sigmoid(z[(p            ) * 16 + m_e]);
            const float gf = fast_sigmoid(z[(p +     UNITS) * 16 + m_e]);
            const float gg = fast_tanh   (z[(p + 2 * UNITS) * 16 + m_e]);
            const float go = fast_sigmoid(z[(p + 3 * UNITS) * 16 + m_e]);
            const int  ci = m_e * UNITS + p;
            const float c = gf * c_st[ci] + gi * gg;
            c_st[ci] = c;
            h_act[m_e * KP + p] = (_Float16)(go * fast_tanh(c));
        }
        if (p_e == 0) {                      // tail column p = 64 -> h_col only
            const int p = 64;
            const float gi = fast_sigmoid(z[(p            ) * 16 + m_e]);
            const float gf = fast_sigmoid(z[(p +     UNITS) * 16 + m_e]);
            const float gg = fast_tanh   (z[(p + 2 * UNITS) * 16 + m_e]);
            const float go = fast_sigmoid(z[(p + 3 * UNITS) * 16 + m_e]);
            const int  ci = m_e * UNITS + p;
            const float c = gf * c_st[ci] + gi * gg;
            c_st[ci] = c;
            h_col[m_e] = go * fast_tanh(c);
        }
    }
    __syncthreads();
}

__global__ __launch_bounds__(NTHREADS) void charrnn_lstm_kernel(
    const float* __restrict__ x,  const float* __restrict__ W,
    const float* __restrict__ U,  const float* __restrict__ b,
    const float* __restrict__ Wd, const float* __restrict__ bd,
    float* __restrict__ out) {
    __shared__ __align__(16) _Float16 s_xin[MT * KP];
    __shared__ __align__(16) _Float16 s_h1[MT * KP];
    __shared__ __align__(16) _Float16 s_h2[MT * KP];
    __shared__ __align__(16) _Float16 s_h3[MT * KP];
    __shared__ __align__(16) float s_xcol[MT];             // column 64 of x_t
    __shared__ __align__(16) float s_hc1[MT];              // column 64 of h1..h3
    __shared__ __align__(16) float s_hc2[MT];
    __shared__ __align__(16) float s_hc3[MT];
    __shared__ __align__(16) float s_c1[MT * UNITS];
    __shared__ __align__(16) float s_c2[MT * UNITS];
    __shared__ __align__(16) float s_c3[MT * UNITS];
    __shared__ __align__(16) float s_z[NZC * MT];          // column-major [n][m]

    const int tid  = threadIdx.x;
    const int lane = tid & 31;
    const int wid  = tid >> 5;                             // 0..8
    const int blk  = blockIdx.x;
    const int m_e  = tid & 15;                             // elementwise row
    const int p_e  = tid >> 4;                             // elementwise col base

    // Persistent weight fragments in VGPRs (same W/U shared by all 3 layers).
    v16h BW[2][2], BU[2][2], BD[2];
    float BB[2], W64[2], U64[2];
#pragma unroll
    for (int j = 0; j < 2; ++j) {
        const int nt = wid + 9 * j;
#pragma unroll
        for (int ks = 0; ks < 2; ++ks) {
            BW[j][ks] = load_bfrag_global(W, NGATE, UNITS, NGATE, nt, ks, lane);
            BU[j][ks] = load_bfrag_global(U, NGATE, UNITS, NGATE, nt, ks, lane);
        }
        const int n = nt * 16 + (lane & 15);
        BB[j]  = (n < NGATE) ? b[n]              : 0.0f;
        W64[j] = (n < NGATE) ? W[64 * NGATE + n] : 0.0f;
        U64[j] = (n < NGATE) ? U[64 * NGATE + n] : 0.0f;
    }
#pragma unroll
    for (int ks = 0; ks < 2; ++ks)
        BD[ks] = load_bfrag_global(Wd, UNITS, UNITS, UNITS, wid, ks, lane);
    const int nd = wid * 16 + (lane & 15);
    const bool dvalid = (wid < NT_D) && (nd < UNITS);
    const float BDv  = dvalid ? bd[nd]              : 0.0f;
    const float WD64 = dvalid ? Wd[64 * UNITS + nd] : 0.0f;

    // Zero-init state / pads.
    for (int i = tid; i < MT * KP; i += NTHREADS) {
        s_xin[i] = (_Float16)0.0f; s_h1[i] = (_Float16)0.0f;
        s_h2[i]  = (_Float16)0.0f; s_h3[i] = (_Float16)0.0f;
    }
    for (int i = tid; i < MT * UNITS; i += NTHREADS) {
        s_c1[i] = 0.0f; s_c2[i] = 0.0f; s_c3[i] = 0.0f;
    }
    if (tid < MT) {
        s_xcol[tid] = 0.0f; s_hc1[tid] = 0.0f;
        s_hc2[tid]  = 0.0f; s_hc3[tid] = 0.0f;
    }
    __syncthreads();

    for (int t = 0; t < SEQ; ++t) {
        // Stage x_t tile -> LDS (f16): waves 0..7 do coalesced 64-wide rows,
        // wave 8 does column 64 + next-step prefetch (global_prefetch_b8).
        if (wid < 8) {
#pragma unroll
            for (int it = 0; it < 4; ++it) {
                const int idx = tid + 256 * it;            // 0..1023
                const int m = idx >> 6, p = idx & 63;
                const int bg = blk * MT + m;
                float v = 0.0f;
                if (bg < BATCH) v = x[((size_t)bg * SEQ + t) * UNITS + p];
                s_xin[m * KP + p] = (_Float16)v;
            }
        } else {
            if (lane < MT) {
                const int bg = blk * MT + lane;
                float v = 0.0f;
                if (bg < BATCH) v = x[((size_t)bg * SEQ + t) * UNITS + 64];
                s_xcol[lane] = v;
            }
            if (t + 1 < SEQ) {
                const int m = lane >> 1;
                const int bg = blk * MT + m;
                if (bg < BATCH)
                    __builtin_prefetch(
                        &x[((size_t)bg * SEQ + (t + 1)) * UNITS + (lane & 1) * 32],
                        0, 1);
            }
        }
        __syncthreads();

        lstm_cell(s_xin, s_xcol, s_h1, s_hc1, s_c1, s_z, BW, BU, BB, W64, U64,
                  wid, lane, m_e, p_e);
        lstm_cell(s_h1,  s_hc1,  s_h2, s_hc2, s_c2, s_z, BW, BU, BB, W64, U64,
                  wid, lane, m_e, p_e);
        lstm_cell(s_h2,  s_hc2,  s_h3, s_hc3, s_c3, s_z, BW, BU, BB, W64, U64,
                  wid, lane, m_e, p_e);

        // Fused dense projection: out_t = h3 @ Wd + bd  (waves 0..4, wave-uniform)
        if (wid < NT_D) {
            v16h A3[2];
#pragma unroll
            for (int ks = 0; ks < 2; ++ks) A3[ks] = load_afrag_lds(s_h3, ks, lane);
            const int mb = (lane >> 4) << 3;
            const v8f h3T = load_col8(s_hc3, mb);
            v8f acc = {BDv, BDv, BDv, BDv, BDv, BDv, BDv, BDv};  // bias in C
#pragma unroll
            for (int ks = 0; ks < 2; ++ks)
                acc = __builtin_amdgcn_wmma_f32_16x16x32_f16(
                    false, A3[ks], false, BD[ks], (short)0, acc, false, false);
            const int n = wid * 16 + (lane & 15);
            if (n < UNITS) {
#pragma unroll
                for (int v = 0; v < 8; ++v) {
                    const float o = __builtin_fmaf(h3T[v], WD64, acc[v]);
                    const int bg = blk * MT + mb + v;
                    if (bg < BATCH)
                        out[((size_t)bg * SEQ + t) * UNITS + n] = o;
                }
            }
        }
        // Next iteration's x-staging barrier orders these h3 reads vs. future writes.
    }
}

extern "C" void kernel_launch(void* const* d_in, const int* in_sizes, int n_in,
                              void* d_out, int out_size, void* d_ws, size_t ws_size,
                              hipStream_t stream) {
    (void)in_sizes; (void)n_in; (void)out_size; (void)d_ws; (void)ws_size;
    const float* x  = (const float*)d_in[0];
    const float* W  = (const float*)d_in[1];
    const float* U  = (const float*)d_in[2];
    const float* b  = (const float*)d_in[3];
    const float* Wd = (const float*)d_in[4];
    const float* bd = (const float*)d_in[5];
    float* out = (float*)d_out;

    dim3 grid((BATCH + MT - 1) / MT);   // 4 batch tiles of 16 rows
    dim3 block(NTHREADS);               // 9 wave32 waves
    hipLaunchKernelGGL(charrnn_lstm_kernel, grid, block, 0, stream,
                       x, W, U, b, Wd, bd, out);
}
